// Multi_Cross_Attention_Layer_13649406066964
// MI455X (gfx1250) — matrix operand order repaired and checked by reading, not compiled, not run on hardware
//
#include <hip/hip_runtime.h>
#include <hip/hip_bf16.h>
#include <stdint.h>

// Problem constants (from reference): S=1024, B=32, D=512
#define S_LEN 1024
#define BATCH 32
#define DMODEL 512

typedef __attribute__((ext_vector_type(16))) __bf16 v16bf;
typedef __attribute__((ext_vector_type(8)))  float  v8f;
typedef __attribute__((ext_vector_type(4)))  int    v4i_t;

union Frag16 {
    v16bf v;
    uint4 u[2];
};

// ---------------------------------------------------------------------------
// gfx1250 async global->LDS copies (ASYNCcnt-tracked), guarded so a missing
// builtin name degrades to a synchronous copy instead of breaking the build.
// Builtin signature (from hipcc diagnostic): first param is int4* in the
// global address space; pass typed vector pointers in AS(1)/AS(3).
// ---------------------------------------------------------------------------
#if __has_builtin(__builtin_amdgcn_global_load_async_to_lds_b128) && \
    __has_builtin(__builtin_amdgcn_s_wait_asynccnt)
#define HAVE_ASYNC_LDS 1
#else
#define HAVE_ASYNC_LDS 0
#endif

__device__ __forceinline__ void async_copy_b128(const void* gsrc, void* ldst) {
#if HAVE_ASYNC_LDS
    // Generic LDS pointer carries the LDS byte offset in its low 32 bits;
    // generic global pointer == global address. Integer casts perform the
    // address-space re-interpretation.
    __builtin_amdgcn_global_load_async_to_lds_b128(
        (__attribute__((address_space(1))) v4i_t*)(uintptr_t)gsrc,
        (__attribute__((address_space(3))) v4i_t*)(uint32_t)(uintptr_t)ldst,
        0, 0);
#else
    *(uint4*)ldst = *(const uint4*)gsrc;
#endif
}

__device__ __forceinline__ void async_wait_all() {
#if HAVE_ASYNC_LDS
    __builtin_amdgcn_s_wait_asynccnt(0);
#endif
}

// ---------------------------------------------------------------------------
// Fragment loaders (CDNA5 WMMA 16x16x32 bf16 layouts, wave32)
// A (16Mx32K): lanes 0-15 -> rows 0-15, K {0..7,16..23}; lanes 16-31 -> rows
// 0-15, K {8..15,24..31}. Contiguous-K in LDS -> two ds_load_b128 per lane.
// ---------------------------------------------------------------------------
__device__ __forceinline__ v16bf load_a_frag(const __bf16* lds, int stride,
                                             int rowBase, int k0, int lane) {
    const int half = lane >> 4;
    const int r = rowBase + (lane & 15);
    const __bf16* p = lds + (size_t)r * stride + k0;
    Frag16 f;
    f.u[0] = *(const uint4*)(p + 8 * half);
    f.u[1] = *(const uint4*)(p + 16 + 8 * half);
    return f.v;
}

// B (32Kx16N): lane n (0-15) -> column n with K {0..7,16..23}; lanes 16-31 ->
// same columns with K {8..15,24..31}. Column c of B == row c of a [N,K] tile.
__device__ __forceinline__ v16bf load_b_frag(const __bf16* lds, int stride,
                                             int colBase, int k0, int lane) {
    const int half = lane >> 4;
    const int c = colBase + (lane & 15);
    const __bf16* p = lds + (size_t)c * stride + k0;
    Frag16 f;
    f.u[0] = *(const uint4*)(p + 8 * half);
    f.u[1] = *(const uint4*)(p + 16 + 8 * half);
    return f.v;
}

__device__ __forceinline__ v8f wmma_bf16(v16bf a, v16bf b, v8f c) {
    return __builtin_amdgcn_wmma_f32_16x16x32_bf16(
        /*neg_a=*/false, a, /*neg_b=*/false, b,
        /*c_mod=*/(short)0, c, /*reuse_a=*/false, /*reuse_b=*/false);
}

// ---------------------------------------------------------------------------
// f32 -> bf16 elementwise convert (for weights)
// ---------------------------------------------------------------------------
__global__ __launch_bounds__(256) void cvt_f32_bf16_kernel(
    const float* __restrict__ src, __bf16* __restrict__ dst, int n4) {
    int i = blockIdx.x * blockDim.x + threadIdx.x;
    if (i < n4) {
        float4 f = ((const float4*)src)[i];
        union { __bf16 b[4]; uint2 u; } t;
        t.b[0] = (__bf16)f.x; t.b[1] = (__bf16)f.y;
        t.b[2] = (__bf16)f.z; t.b[3] = (__bf16)f.w;
        ((uint2*)dst)[i] = t.u;
    }
}

// ---------------------------------------------------------------------------
// GEMM: C[M,N] = X[M,K] * W[N,K]^T   (torch Linear convention, row-major)
// Block = 256 threads = 8 waves (4 M-waves x 2 N-waves), tile 128x64.
// Each wave: 32x32 output = 2x2 WMMA accumulators. K staged 32-wide in LDS,
// double-buffered: async prefetch of tile k+1 overlaps WMMA on tile k, one
// barrier per k-step.
// ---------------------------------------------------------------------------
template <bool X_BF16, bool OUT_F32_BIAS>
__global__ __launch_bounds__(256) void gemm_wmma_kernel(
    const void* __restrict__ Xv, const __bf16* __restrict__ W,
    const float* __restrict__ bias, void* __restrict__ Out,
    int M, int N, int K) {
    constexpr int LSTR = 40;  // padded LDS row stride (80B, 16B-aligned frags)
    __shared__ __bf16 Xs[2][128 * LSTR];
    __shared__ __bf16 Ws[2][64 * LSTR];

    const int tid = threadIdx.x;
    const int lane = tid & 31;
    const int wave = tid >> 5;
    const int wm = wave >> 1;  // 0..3
    const int wn = wave & 1;   // 0..1
    const int m0 = blockIdx.x * 128;
    const int n0 = blockIdx.y * 64;

    const int row = tid >> 1;   // staging row (X: 0..127, W: 0..63)
    const int half = tid & 1;   // 16-element half of the 32-wide K slice

    auto stage = [&](int buf, int k0) {
        __bf16* xdst = &Xs[buf][row * LSTR + 16 * half];
        if constexpr (X_BF16) {
            const __bf16* xp =
                (const __bf16*)Xv + (size_t)(m0 + row) * K + k0 + 16 * half;
            async_copy_b128(xp, xdst);
            async_copy_b128(xp + 8, xdst + 8);
        } else {
            const float* xp =
                (const float*)Xv + (size_t)(m0 + row) * K + k0 + 16 * half;
            Frag16 t;
#pragma unroll
            for (int i = 0; i < 16; i++) t.v[i] = (__bf16)xp[i];
            ((uint4*)xdst)[0] = t.u[0];
            ((uint4*)xdst)[1] = t.u[1];
        }
        if (tid < 128) {
            const __bf16* wp = W + (size_t)(n0 + row) * K + k0 + 16 * half;
            __bf16* wdst = &Ws[buf][row * LSTR + 16 * half];
            async_copy_b128(wp, wdst);
            async_copy_b128(wp + 8, wdst + 8);
        }
    };

    const v8f vzero = {0.f, 0.f, 0.f, 0.f, 0.f, 0.f, 0.f, 0.f};
    v8f acc[2][2];
#pragma unroll
    for (int i = 0; i < 2; i++)
#pragma unroll
        for (int j = 0; j < 2; j++) acc[i][j] = vzero;

    stage(0, 0);
    async_wait_all();
    __syncthreads();

    const int nk = K / 32;
    for (int ki = 0; ki < nk; ki++) {
        const int cur = ki & 1;
        if (ki + 1 < nk) stage(cur ^ 1, (ki + 1) * 32);

        v16bf A[2], Bf[2];
#pragma unroll
        for (int i = 0; i < 2; i++)
            A[i] = load_a_frag(&Xs[cur][0], LSTR, wm * 32 + i * 16, 0, lane);
#pragma unroll
        for (int j = 0; j < 2; j++)
            Bf[j] = load_b_frag(&Ws[cur][0], LSTR, wn * 32 + j * 16, 0, lane);
#pragma unroll
        for (int i = 0; i < 2; i++)
#pragma unroll
            for (int j = 0; j < 2; j++)
                acc[i][j] = wmma_bf16(A[i], Bf[j], acc[i][j]);

        async_wait_all();
        __syncthreads();
    }

    // ---- epilogue: C-frag layout -> memory ----
    const int mhalf = (lane < 16) ? 0 : 8;
    const int ncol = lane & 15;
#pragma unroll
    for (int i = 0; i < 2; i++) {
#pragma unroll
        for (int j = 0; j < 2; j++) {
#pragma unroll
            for (int v = 0; v < 8; v++) {
                const int m = m0 + wm * 32 + i * 16 + v + mhalf;
                const int n = n0 + wn * 32 + j * 16 + ncol;
                const float val = acc[i][j][v];
                if constexpr (OUT_F32_BIAS) {
                    ((float*)Out)[(size_t)m * N + n] = val + bias[n];
                } else {
                    ((__bf16*)Out)[(size_t)m * N + n] = (__bf16)val;
                }
            }
        }
    }
}

// ---------------------------------------------------------------------------
// Flash attention for one (stream, batch, 64-query tile).
// Block = 128 threads = 4 waves; each wave owns 16 query rows with full
// f32 accumulators (16x512 -> 32 v8f per wave). Keys processed 32 at a time;
// K and transposed-V tiles are double-buffered so staging of tile kt+1
// overlaps the WMMA/softmax work on tile kt.
// Q/K/V are bf16 buffers laid out [S, B, D] (row index s*B + b).
// Output written as bf16 into concat buffer [S, B, 2D] at column colOff.
// ---------------------------------------------------------------------------
__global__ __launch_bounds__(128) void flash_attn_kernel(
    const __bf16* __restrict__ Q, const __bf16* __restrict__ Kb,
    const __bf16* __restrict__ Vb, __bf16* __restrict__ Out, int colOff) {
    constexpr int QT = 64;            // query rows per block
    constexpr int KT = 32;            // keys per iteration
    constexpr int QSTR = DMODEL + 8;  // 520
    constexpr int KSTR = DMODEL + 8;  // 520
    constexpr int VSTR = 40;          // transposed-V row stride
    constexpr int PSTR = 40;          // per-wave probability tile stride

    __shared__ __bf16 Qs[QT * QSTR];          //  66,560 B
    __shared__ __bf16 Ks[2][KT * KSTR];       //  66,560 B
    __shared__ __bf16 Vt[2][DMODEL * VSTR];   //  81,920 B (V transposed [e][n])
    __shared__ __bf16 Ps[4 * 16 * PSTR];      //   5,120 B

    const int tid = threadIdx.x;
    const int lane = tid & 31;
    const int wave = tid >> 5;  // 0..3, owns query rows wave*16..wave*16+15
    const int b = blockIdx.y;
    const int s0 = blockIdx.x * QT;

    // ---- stage K (async, row-major) + V (transposed scatter) for tile kt ----
    auto stage_kv = [&](int buf, int kt) {
        {
            const int krow = tid >> 2;  // 0..31
            const int seg = tid & 3;    // 128-element quarter of the row
            const __bf16* src =
                Kb + ((size_t)(kt * KT + krow) * BATCH + b) * DMODEL + seg * 128;
            __bf16* dst = &Ks[buf][krow * KSTR + seg * 128];
#pragma unroll
            for (int i = 0; i < 16; i++)
                async_copy_b128(src + i * 8, dst + i * 8);
        }
        {
            const int n = tid >> 2;   // key index 0..31
            const int seg = tid & 3;  // 128-element quarter of the feature dim
            const uint4* src =
                (const uint4*)(Vb + ((size_t)(kt * KT + n) * BATCH + b) * DMODEL +
                               seg * 128);
#pragma unroll
            for (int c = 0; c < 16; c++) {
                uint4 blk = src[c];
                const __bf16* pb = (const __bf16*)&blk;
#pragma unroll
                for (int e = 0; e < 8; e++)
                    Vt[buf][(seg * 128 + c * 8 + e) * VSTR + n] = pb[e];
            }
        }
    };

    // ---- stage Q tile (async): 64 rows x 512, 2 threads per row ----
    {
        const int r = tid >> 1;
        const int part = tid & 1;
        const __bf16* src =
            Q + ((size_t)(s0 + r) * BATCH + b) * DMODEL + part * 256;
        __bf16* dst = Qs + r * QSTR + part * 256;
#pragma unroll
        for (int i = 0; i < 32; i++) async_copy_b128(src + i * 8, dst + i * 8);
    }

    const v8f vzero = {0.f, 0.f, 0.f, 0.f, 0.f, 0.f, 0.f, 0.f};
    v8f oacc[32];
#pragma unroll
    for (int t = 0; t < 32; t++) oacc[t] = vzero;
    float rmax[8], rsum[8];
#pragma unroll
    for (int v = 0; v < 8; v++) {
        rmax[v] = -__builtin_inff();
        rsum[v] = 0.0f;
    }

    stage_kv(0, 0);
    async_wait_all();
    __syncthreads();

    constexpr int NT = S_LEN / KT;  // 32 key tiles
    for (int kt = 0; kt < NT; kt++) {
        const int cur = kt & 1;
        if (kt + 1 < NT) stage_kv(cur ^ 1, kt + 1);

        // ---- scores: S[16 x 32] = Q_tile * K_tile^T over D=512 ----
        v8f sc[2];
        sc[0] = vzero;
        sc[1] = vzero;
#pragma unroll
        for (int ks = 0; ks < DMODEL / 32; ks++) {
            v16bf A = load_a_frag(Qs, QSTR, wave * 16, ks * 32, lane);
#pragma unroll
            for (int j = 0; j < 2; j++) {
                v16bf Bf = load_b_frag(&Ks[cur][0], KSTR, j * 16, ks * 32, lane);
                sc[j] = wmma_bf16(A, Bf, sc[j]);
            }
        }

        // ---- online softmax (rows live in 16-lane halves of the C-frag) ----
        float corr[8];
#pragma unroll
        for (int v = 0; v < 8; v++) {
            const float sv0 = sc[0][v];
            const float sv1 = sc[1][v];
            float mx = fmaxf(sv0, sv1);
#pragma unroll
            for (int off = 1; off < 16; off <<= 1)
                mx = fmaxf(mx, __shfl_xor(mx, off, 32));
            const float mnew = fmaxf(rmax[v], mx);
            const float c = __expf(rmax[v] - mnew);
            const float p0 = __expf(sv0 - mnew);
            const float p1 = __expf(sv1 - mnew);
            float ps = p0 + p1;
#pragma unroll
            for (int off = 1; off < 16; off <<= 1) ps += __shfl_xor(ps, off, 32);
            rsum[v] = rsum[v] * c + ps;
            rmax[v] = mnew;
            corr[v] = c;
            // P (C-frag layout) -> per-wave LDS tile in row-major [m][n]
            const int mr = v + ((lane < 16) ? 0 : 8);
            __bf16* pp = Ps + (wave * 16 + mr) * PSTR;
            pp[lane & 15] = (__bf16)p0;
            pp[16 + (lane & 15)] = (__bf16)p1;
        }

        // rescale existing output accumulators (corr matches row mapping)
#pragma unroll
        for (int t = 0; t < 32; t++)
#pragma unroll
            for (int v = 0; v < 8; v++) oacc[t][v] *= corr[v];

        // ---- O += P[16x32] * V[32x512] ----
        v16bf Pf = load_a_frag(Ps, PSTR, wave * 16, 0, lane);
#pragma unroll
        for (int t = 0; t < 32; t++) {
            v16bf Bf = load_b_frag(&Vt[cur][0], VSTR, t * 16, 0, lane);
            oacc[t] = wmma_bf16(Pf, Bf, oacc[t]);
        }

        async_wait_all();
        __syncthreads();
    }

    // ---- normalize and write bf16 into concat buffer [S, B, 2D] ----
    const int mhalf = (lane < 16) ? 0 : 8;
    const int nc = lane & 15;
#pragma unroll
    for (int v = 0; v < 8; v++) {
        const float inv = 1.0f / rsum[v];
        const int srow = s0 + wave * 16 + v + mhalf;
        __bf16* orow = Out + ((size_t)srow * BATCH + b) * (2 * DMODEL) + colOff;
#pragma unroll
        for (int t = 0; t < 32; t++)
            orow[t * 16 + nc] = (__bf16)(oacc[t][v] * inv);
    }
}

// ---------------------------------------------------------------------------
// Host-side orchestration
// d_in: inputs_t, inputs_d, Wqt, Wkt, Wvt, Wqd, Wkd, Wvd, Wout, bout
// d_out: [S, B, D] f32
// ---------------------------------------------------------------------------
extern "C" void kernel_launch(void* const* d_in, const int* in_sizes, int n_in,
                              void* d_out, int out_size, void* d_ws,
                              size_t ws_size, hipStream_t stream) {
    const float* in_t = (const float*)d_in[0];
    const float* in_d = (const float*)d_in[1];
    const float* Wsrc[7] = {(const float*)d_in[2], (const float*)d_in[3],
                            (const float*)d_in[4], (const float*)d_in[5],
                            (const float*)d_in[6], (const float*)d_in[7],
                            (const float*)d_in[8]};
    const float* bout = (const float*)d_in[9];

    constexpr size_t WQKV_ELEMS = (size_t)DMODEL * DMODEL;        // 262,144
    constexpr size_t WOUT_ELEMS = (size_t)DMODEL * 2 * DMODEL;    // 524,288
    constexpr size_t QKV_ELEMS = (size_t)S_LEN * BATCH * DMODEL;  // 16,777,216

    __bf16* wbf = (__bf16*)d_ws;                      // 6 QKV weights + Wout
    __bf16* qkv = wbf + 6 * WQKV_ELEMS + WOUT_ELEMS;  // Qt,Kt,Vt,Qd,Kd,Vd
    __bf16* attn = qkv + 6 * QKV_ELEMS;               // concat attention out
    __bf16* woutbf = wbf + 6 * WQKV_ELEMS;

    // 1) convert weights to bf16
    for (int i = 0; i < 6; i++) {
        int n4 = (int)(WQKV_ELEMS / 4);
        cvt_f32_bf16_kernel<<<(n4 + 255) / 256, 256, 0, stream>>>(
            Wsrc[i], wbf + i * WQKV_ELEMS, n4);
    }
    {
        int n4 = (int)(WOUT_ELEMS / 4);
        cvt_f32_bf16_kernel<<<(n4 + 255) / 256, 256, 0, stream>>>(
            Wsrc[6], woutbf, n4);
    }

    // 2) QKV projections: rows are raw [S,B] rows (s*B + b), K = D = 512
    const int M = S_LEN * BATCH;
    dim3 gg(M / 128, DMODEL / 64);
    for (int z = 0; z < 6; z++) {
        const void* X = (z < 3) ? (const void*)in_t : (const void*)in_d;
        gemm_wmma_kernel<false, false><<<gg, 256, 0, stream>>>(
            X, wbf + z * WQKV_ELEMS, nullptr, qkv + (size_t)z * QKV_ELEMS,
            M, DMODEL, DMODEL);
    }

    const __bf16* Qt = qkv + 0 * QKV_ELEMS;
    const __bf16* Kt = qkv + 1 * QKV_ELEMS;
    const __bf16* Vt = qkv + 2 * QKV_ELEMS;
    const __bf16* Qd = qkv + 3 * QKV_ELEMS;
    const __bf16* Kd = qkv + 4 * QKV_ELEMS;
    const __bf16* Vd = qkv + 5 * QKV_ELEMS;

    // 3) fused attention: stream "t": softmax(Q_d K_t^T) V_t -> cols [0,512)
    //                     stream "d": softmax(Q_t K_d^T) V_d -> cols [512,1024)
    dim3 ga(S_LEN / 64, BATCH);
    flash_attn_kernel<<<ga, 128, 0, stream>>>(Qd, Kt, Vt, attn, 0);
    flash_attn_kernel<<<ga, 128, 0, stream>>>(Qt, Kd, Vd, attn, DMODEL);

    // 4) output projection: [M, 2D] * Wout[D, 2D]^T + bout -> f32 [S,B,D]
    gemm_wmma_kernel<true, true><<<gg, 256, 0, stream>>>(
        attn, woutbf, bout, d_out, M, DMODEL, 2 * DMODEL);
}
